// StarPoolOutputModel_39513699123757
// MI455X (gfx1250) — compile-verified, matrix operand order checked by reading, not run on hardware
//
#include <hip/hip_runtime.h>
#include <hip/hip_bf16.h>

typedef __attribute__((ext_vector_type(16))) __bf16 v16bf;
typedef __attribute__((ext_vector_type(8)))  float  v8f;

#define HID 128
#define ECH 64
#define DEG 16

// K index inside a 16x32 bf16 WMMA operand tile for element j (0..15) of a lane.
// Lanes 0-15: K = {0..7, 16..23}; lanes 16-31: K = {8..15, 24..31}.
__device__ __host__ __forceinline__ int kmap(int j, int lane_hi) {
    int b0 = lane_hi ? 8 : 0;
    return (j < 8) ? (b0 + j) : (16 + b0 + (j - 8));
}

// ---------------------------------------------------------------------------
// K0: pack weights (row-major fp32 [K][N]) into bf16 B-operand layout:
//     packed[ntile][kstep][lane][j]  (each lane reads 16 contiguous bf16 = 32B)
//     Also zero the per-graph accumulators.
// ---------------------------------------------------------------------------
__global__ __launch_bounds__(256) void k_prep(const float* __restrict__ Wq,
                                              const float* __restrict__ Wkv,
                                              const float* __restrict__ Wdk,
                                              __bf16* __restrict__ Wq_p,
                                              __bf16* __restrict__ Wkv_p,
                                              __bf16* __restrict__ Wdk_p,
                                              float* __restrict__ gsum,
                                              float* __restrict__ gcnt) {
    int t = blockIdx.x * blockDim.x + threadIdx.x;
    if (t < 8 * 4 * 32 * 16) {  // Wq / Wkv: K=128 (4 ksteps), N=128 (8 ntiles)
        int j = t & 15, lane = (t >> 4) & 31, kb = (t >> 9) & 3, nt = t >> 11;
        int K = kb * 32 + kmap(j, lane >> 4);
        int N = nt * 16 + (lane & 15);
        Wq_p[t]  = (__bf16)Wq[K * HID + N];
        Wkv_p[t] = (__bf16)Wkv[K * HID + N];
    }
    if (t < 8 * 2 * 32 * 16) {  // Wdk: K=64 (2 ksteps), N=128 (8 ntiles)
        int j = t & 15, lane = (t >> 4) & 31, kb = (t >> 9) & 1, nt = t >> 10;
        int K = kb * 32 + kmap(j, lane >> 4);
        int N = nt * 16 + (lane & 15);
        Wdk_p[t] = (__bf16)Wdk[K * HID + N];
    }
    if (t < 64 * HID) gsum[t] = 0.f;
    if (t < 64) gcnt[t] = 0.f;
}

// ---------------------------------------------------------------------------
// K1: per 16-node tile (one wave32): layernorm(x) -> q = xn@Wq+bq, k = xn@Wkv+bkv
//     B fragments software-pipelined (one step of lookahead); D tiles staged
//     through LDS so global stores are coalesced float4 rows.
// ---------------------------------------------------------------------------
__global__ __launch_bounds__(32) void k_node_ln_qk(const float* __restrict__ x,
                                                   const float* __restrict__ bq,
                                                   const float* __restrict__ bkv,
                                                   const float* __restrict__ ln_g,
                                                   const float* __restrict__ ln_b,
                                                   const __bf16* __restrict__ Wq_p,
                                                   const __bf16* __restrict__ Wkv_p,
                                                   float* __restrict__ q_out,
                                                   float* __restrict__ k_out,
                                                   int n) {
    __shared__ float xs[16][HID];
    __shared__ float qs[16][HID];
    __shared__ float ks[16][HID];
    __shared__ float lg[HID], lb[HID], bqs[HID], bks[HID];
    __shared__ float mu[16], rs[16];
    const int lane = threadIdx.x;
    const int base = blockIdx.x * 16;
    const bool full = (base + 16 <= n);

    // Coalesced staging of params (each lane: 4 contiguous floats per array).
    *(float4*)&lg[lane * 4]  = *(const float4*)(ln_g + lane * 4);
    *(float4*)&lb[lane * 4]  = *(const float4*)(ln_b + lane * 4);
    *(float4*)&bqs[lane * 4] = *(const float4*)(bq + lane * 4);
    *(float4*)&bks[lane * 4] = *(const float4*)(bkv + lane * 4);
    if (full) {
        for (int r = 0; r < 16; ++r)
            *(float4*)&xs[r][lane * 4] = *(const float4*)(x + (size_t)(base + r) * HID + lane * 4);
    } else {
        for (int r = 0; r < 16; ++r) {
            float4 v = (base + r < n) ? *(const float4*)(x + (size_t)(base + r) * HID + lane * 4)
                                      : make_float4(0.f, 0.f, 0.f, 0.f);
            *(float4*)&xs[r][lane * 4] = v;
        }
    }
    __syncthreads();
    if (lane < 16) {
        float s = 0.f, s2 = 0.f;
        for (int c = 0; c < HID; ++c) { float v = xs[lane][c]; s += v; s2 += v * v; }
        float m = s * (1.f / HID);
        mu[lane] = m;
        rs[lane] = rsqrtf(s2 * (1.f / HID) - m * m + 1e-5f);
    }
    __syncthreads();

    const int M = lane & 15;
    const int hi = lane >> 4;
    const float m = mu[M], r = rs[M];

    v16bf A[4];
    for (int kb = 0; kb < 4; ++kb)
        for (int j = 0; j < 16; ++j) {
            int K = kb * 32 + kmap(j, hi);
            float v = (xs[M][K] - m) * r * lg[K] + lb[K];
            A[kb][j] = (__bf16)v;
        }

    // Flattened GEMM loop: step = nt*4 + kb, one step of B lookahead.
    auto ldB = [&](const __bf16* p, int idx) {
        return *(const v16bf*)(p + ((size_t)idx * 32 + lane) * 16);
    };
    v16bf Bq = ldB(Wq_p, 0), Bk = ldB(Wkv_p, 0);
    v8f cq, ck;
#pragma unroll
    for (int step = 0; step < 32; ++step) {
        const int nt = step >> 2, kb = step & 3;
        if (kb == 0) {
            float biasq = bqs[nt * 16 + M], biask = bks[nt * 16 + M];
            for (int v = 0; v < 8; ++v) { cq[v] = biasq; ck[v] = biask; }
        }
        v16bf Bq_n, Bk_n;
        if (step < 31) { Bq_n = ldB(Wq_p, step + 1); Bk_n = ldB(Wkv_p, step + 1); }
        cq = __builtin_amdgcn_wmma_f32_16x16x32_bf16(false, A[kb], false, Bq,
                                                     (short)0, cq, false, false);
        ck = __builtin_amdgcn_wmma_f32_16x16x32_bf16(false, A[kb], false, Bk,
                                                     (short)0, ck, false, false);
        Bq = Bq_n; Bk = Bk_n;
        if (kb == 3) {
            for (int v = 0; v < 8; ++v) {  // D: row = v + 8*hi, col = nt*16+M
                qs[v + 8 * hi][nt * 16 + M] = cq[v];
                ks[v + 8 * hi][nt * 16 + M] = ck[v];
            }
        }
    }
    __syncthreads();

    // Coalesced float4 row stores.
    if (full) {
        for (int rr = 0; rr < 16; ++rr) {
            *(float4*)(q_out + (size_t)(base + rr) * HID + lane * 4) = *(float4*)&qs[rr][lane * 4];
            *(float4*)(k_out + (size_t)(base + rr) * HID + lane * 4) = *(float4*)&ks[rr][lane * 4];
        }
    } else {
        for (int rr = 0; rr < 16; ++rr) {
            if (base + rr < n) {
                *(float4*)(q_out + (size_t)(base + rr) * HID + lane * 4) = *(float4*)&qs[rr][lane * 4];
                *(float4*)(k_out + (size_t)(base + rr) * HID + lane * 4) = *(float4*)&ks[rr][lane * 4];
            }
        }
    }
}

// ---------------------------------------------------------------------------
// K2: per node (one wave32): dk GEMM (16 edges x 64 -> 128), silu attention,
//     scatter-mean (16 contiguous edges/node), residual + LN + residual,
//     per-graph atomic accumulation. Wave-level reductions via __shfl_xor.
// ---------------------------------------------------------------------------
__global__ __launch_bounds__(32) void k_edge(const float* __restrict__ x,
                                             const float* __restrict__ edge_attr,
                                             const float* __restrict__ bdk,
                                             const __bf16* __restrict__ Wdk_p,
                                             const float* __restrict__ q_ws,
                                             const float* __restrict__ k_ws,
                                             const int* __restrict__ edge_index,
                                             const int* __restrict__ batch,
                                             const float* __restrict__ lo_g,
                                             const float* __restrict__ lo_b,
                                             float* __restrict__ gsum,
                                             float* __restrict__ gcnt,
                                             int n) {
    __shared__ float ea[DEG][ECH];
    __shared__ float dk[DEG][HID];
    __shared__ float kj[DEG][HID];
    __shared__ float qv[HID];
    __shared__ float attn[DEG][32];
    const int lane = threadIdx.x;
    const int i = blockIdx.x;
    const size_t e0 = (size_t)i * DEG;

    // All 16 source indices up front (uniform addresses -> scalar loads).
    int idx[DEG];
    *(int4*)&idx[0]  = *(const int4*)(edge_index + e0);
    *(int4*)&idx[4]  = *(const int4*)(edge_index + e0 + 4);
    *(int4*)&idx[8]  = *(const int4*)(edge_index + e0 + 8);
    *(int4*)&idx[12] = *(const int4*)(edge_index + e0 + 12);

    // Stage edge_attr tile (16x64), q[i], gathered k[src] rows.
    for (int t = 0; t < 8; ++t) {
        int idx4 = t * 32 + lane;
        int r = idx4 >> 4, c4 = idx4 & 15;
        *(float4*)&ea[r][c4 * 4] = *(const float4*)(edge_attr + (e0 + r) * ECH + c4 * 4);
    }
    *(float4*)&qv[lane * 4] = *(const float4*)(q_ws + (size_t)i * HID + lane * 4);
#pragma unroll
    for (int r = 0; r < DEG; ++r)
        *(float4*)&kj[r][lane * 4] = *(const float4*)(k_ws + (size_t)idx[r] * HID + lane * 4);

    // Residual channels owned by this lane (ch = lane*4 + d).
    const float4 xres = *(const float4*)(x + (size_t)i * HID + lane * 4);
    __syncthreads();

    const int M = lane & 15;
    const int hi = lane >> 4;
    v16bf A[2];
    for (int kb = 0; kb < 2; ++kb)
        for (int j = 0; j < 16; ++j)
            A[kb][j] = (__bf16)ea[M][kb * 32 + kmap(j, hi)];

    // Flattened GEMM: step = nt*2 + kb, one step of B lookahead.
    auto ldB = [&](int sidx) {
        return *(const v16bf*)(Wdk_p + ((size_t)sidx * 32 + lane) * 16);
    };
    v16bf B = ldB(0);
    v8f c;
#pragma unroll
    for (int step = 0; step < 16; ++step) {
        const int nt = step >> 1, kb = step & 1;
        if (kb == 0) {
            float bias = bdk[nt * 16 + M];
            for (int v = 0; v < 8; ++v) c[v] = bias;
        }
        v16bf B_n;
        if (step < 15) B_n = ldB(step + 1);
        c = __builtin_amdgcn_wmma_f32_16x16x32_bf16(false, A[kb], false, B,
                                                    (short)0, c, false, false);
        B = B_n;
        if (kb == 1)
            for (int v = 0; v < 8; ++v) dk[v + 8 * hi][nt * 16 + M] = c[v];
    }
    __syncthreads();

    // attn[e][h], lane == head h (NH == 32 == wave width)
    for (int e = 0; e < DEG; ++e) {
        float s = 0.f;
        for (int d = 0; d < 4; ++d) {
            int ch = lane * 4 + d;
            s += qv[ch] * kj[e][ch] * dk[e][ch];
        }
        attn[e][lane] = s / (1.f + __expf(-s));  // silu
    }
    __syncthreads();

    // agg (mean over 16 edges) + residual; lane owns channels lane*4..lane*4+3
    const float xr[4] = {xres.x, xres.y, xres.z, xres.w};
    float hv[4];
    for (int d = 0; d < 4; ++d) {
        int ch = lane * 4 + d;
        float a = 0.f;
        for (int e = 0; e < DEG; ++e) a += kj[e][ch] * attn[e][lane];
        hv[d] = xr[d] + a * (1.f / DEG);
    }
    // layernorm over 128 channels: wave32 butterfly reductions
    float s = hv[0] + hv[1] + hv[2] + hv[3];
    for (int off = 16; off > 0; off >>= 1) s += __shfl_xor(s, off);
    const float mean = s * (1.f / HID);
    float s2 = 0.f;
    for (int d = 0; d < 4; ++d) { float dd = hv[d] - mean; s2 += dd * dd; }
    for (int off = 16; off > 0; off >>= 1) s2 += __shfl_xor(s2, off);
    const float rstd = rsqrtf(s2 * (1.f / HID) + 1e-5f);

    const int b = batch[i];
    for (int d = 0; d < 4; ++d) {
        int ch = lane * 4 + d;
        float hfin = xr[d] + (hv[d] - mean) * rstd * lo_g[ch] + lo_b[ch];
        atomicAdd(&gsum[(size_t)b * HID + ch], hfin);
    }
    if (lane == 0) atomicAdd(&gcnt[b], 1.0f);
}

// ---------------------------------------------------------------------------
// K3: per-graph mean readout + output projection (HID -> 1)
// ---------------------------------------------------------------------------
__global__ __launch_bounds__(128) void k_readout(const float* __restrict__ gsum,
                                                 const float* __restrict__ gcnt,
                                                 const float* __restrict__ Wout,
                                                 const float* __restrict__ bout,
                                                 float* __restrict__ out) {
    __shared__ float red[HID];
    const int b = blockIdx.x, t = threadIdx.x;
    float cnt = gcnt[b];
    if (cnt < 1.f) cnt = 1.f;
    red[t] = gsum[(size_t)b * HID + t] / cnt * Wout[t];
    __syncthreads();
    for (int s = 64; s > 0; s >>= 1) {
        if (t < s) red[t] += red[t + s];
        __syncthreads();
    }
    if (t == 0) out[b] = red[0] + bout[0];
}

// ---------------------------------------------------------------------------
extern "C" void kernel_launch(void* const* d_in, const int* in_sizes, int n_in,
                              void* d_out, int out_size, void* d_ws, size_t ws_size,
                              hipStream_t stream) {
    const float* x        = (const float*)d_in[0];
    const float* edge_attr= (const float*)d_in[1];
    const float* Wq       = (const float*)d_in[2];
    const float* bq       = (const float*)d_in[3];
    const float* Wkv      = (const float*)d_in[4];
    const float* bkv      = (const float*)d_in[5];
    const float* Wdk      = (const float*)d_in[6];
    const float* bdk      = (const float*)d_in[7];
    const float* ln_in_g  = (const float*)d_in[8];
    const float* ln_in_b  = (const float*)d_in[9];
    const float* ln_out_g = (const float*)d_in[10];
    const float* ln_out_b = (const float*)d_in[11];
    const float* Wout     = (const float*)d_in[12];
    const float* bout     = (const float*)d_in[13];
    const int*   edge_index = (const int*)d_in[14];
    const int*   batch    = (const int*)d_in[15];

    const int n = in_sizes[0] / HID;

    char* ws = (char*)d_ws;
    size_t off = 0;
    auto alloc = [&](size_t bytes) {
        void* p = ws + off;
        off = (off + bytes + 255) & ~(size_t)255;
        return p;
    };
    float*  q_ws  = (float*)alloc((size_t)n * HID * sizeof(float));
    float*  k_ws  = (float*)alloc((size_t)n * HID * sizeof(float));
    __bf16* Wq_p  = (__bf16*)alloc(8 * 4 * 32 * 16 * sizeof(__bf16));
    __bf16* Wkv_p = (__bf16*)alloc(8 * 4 * 32 * 16 * sizeof(__bf16));
    __bf16* Wdk_p = (__bf16*)alloc(8 * 2 * 32 * 16 * sizeof(__bf16));
    float*  gsum  = (float*)alloc(64 * HID * sizeof(float));
    float*  gcnt  = (float*)alloc(64 * sizeof(float));

    k_prep<<<64, 256, 0, stream>>>(Wq, Wkv, Wdk, Wq_p, Wkv_p, Wdk_p, gsum, gcnt);
    k_node_ln_qk<<<(n + 15) / 16, 32, 0, stream>>>(x, bq, bkv, ln_in_g, ln_in_b,
                                                   Wq_p, Wkv_p, q_ws, k_ws, n);
    k_edge<<<n, 32, 0, stream>>>(x, edge_attr, bdk, Wdk_p, q_ws, k_ws,
                                 edge_index, batch, ln_out_g, ln_out_b,
                                 gsum, gcnt, n);
    k_readout<<<64, 128, 0, stream>>>(gsum, gcnt, Wout, bout, (float*)d_out);
}
